// DecoderLSTM_257698038395
// MI455X (gfx1250) — compile-verified, matrix operand order checked by reading, not run on hardware
//
#include <hip/hip_runtime.h>
#include <stdint.h>

// ---------- types ----------
typedef __attribute__((ext_vector_type(16))) __bf16 v16bf;
typedef __attribute__((ext_vector_type(8)))  float  v8f;

#define BB    16384
#define HH    512
#define TSTEP 84
#define FOURH 2048
#define W0C   544                 // layer-0 augmented K: 512 h + 3 x + 29 zero pad
#define K0    17                  // layer-0 K tiles (of 32)
#define K1    16                  // layer-1 / projection K tiles
#define AST   552                 // LDS h-row stride (bf16 elems), bank-staggered
#define NW0   (FOURH*W0C)
#define NW    (FOURH*HH)

#define OFF_WHH0 0
#define OFF_WIH1 (NW0)
#define OFF_WHH1 (NW0 + NW)
#define OFF_WPC  (NW0 + 2*NW)
#define WS_ELEMS (NW0 + 2*NW + 16*HH)

#define LOG2E 1.4426950408889634f

__device__ __forceinline__ unsigned short f2bf(float x){
  unsigned u = __float_as_uint(x);
  u += 0x7FFFu + ((u >> 16) & 1u);            // round-to-nearest-even
  return (unsigned short)(u >> 16);
}
// fast sigmoid: rcp(1+exp2(-x*log2e)); overflow-safe (rcp(inf)=0)
__device__ __forceinline__ float fast_sigm(float x){
  float e = __builtin_amdgcn_exp2f(-LOG2E * x);
  return __builtin_amdgcn_rcpf(1.0f + e);
}
// fast tanh: (1-e)/(1+e), e=exp(-2x); clamp to avoid inf/inf
__device__ __forceinline__ float fast_tanh(float x){
  float xc = fminf(fmaxf(x, -15.0f), 15.0f);
  float e  = __builtin_amdgcn_exp2f(-2.0f*LOG2E * xc);
  return (1.0f - e) * __builtin_amdgcn_rcpf(1.0f + e);
}

union FragU { v16bf v; uint4 u[2]; };

// 16-bit A/B fragment: two 16-B chunks at +0 and +16 elements (ISA 7.12.2).
// `off` unsigned 32-bit so loads lower to saddr + voffset (+const IOFFSET).
__device__ __forceinline__ v16bf ldfrag(const unsigned short* base, unsigned off){
  FragU t;
  t.u[0] = *(const uint4*)(base + off);
  t.u[1] = *(const uint4*)(base + off + 16u);
  return t.v;
}

// gate/tile column starts, idx = gate*2 + tile
__device__ __constant__ const unsigned GOFF[8] = {0,16,512,528,1024,1040,1536,1552};

// ---------- prep: stage bf16 weights (Whh0 augmented with Wih0 columns) ----------
__global__ void prep_weights(const float* __restrict__ whh0,
                             const float* __restrict__ wih0,
                             const float* __restrict__ wih1,
                             const float* __restrict__ whh1,
                             const float* __restrict__ wpc,
                             unsigned short* __restrict__ ws){
  int i = blockIdx.x*blockDim.x + threadIdx.x;
  if (i < NW0){
    int n = i / W0C, c = i % W0C;
    float v = (c < HH) ? whh0[n*HH + c]
            : ((c < HH+3) ? wih0[n*3 + (c-HH)] : 0.0f);
    ws[OFF_WHH0 + i] = f2bf(v);
  } else if (i < NW0 + NW){
    ws[i] = f2bf(wih1[i - NW0]);
  } else if (i < NW0 + 2*NW){
    ws[i] = f2bf(whh1[i - (NW0 + NW)]);
  } else if (i < WS_ELEMS){
    int j = i - (NW0 + 2*NW); int r = j >> 9, c = j & 511;
    ws[i] = f2bf(r < 3 ? wpc[r*HH + c] : 0.0f);   // Wpc padded to 16 rows
  }
}

// ---------- persistent LSTM: one WG = 16 batch rows for all 84 steps ----------
__global__ __launch_bounds__(512, 1)
void lstm_persist(const int*   __restrict__ seqlen,
                  const float* __restrict__ bih0, const float* __restrict__ bhh0,
                  const float* __restrict__ bih1, const float* __restrict__ bhh1,
                  const float* __restrict__ bpc,
                  const unsigned short* __restrict__ ws,
                  float* __restrict__ out)          // [B][T][3]
{
  __shared__ unsigned short h0buf[16*AST];  // h0 state + x feedback in cols 512..514
  __shared__ unsigned short h1buf[16*AST];  // h1 state
  __shared__ float lb0[FOURH];              // b_ih0 + b_hh0
  __shared__ float lb1[FOURH];              // b_ih1 + b_hh1

  const unsigned short* whh0 = ws + OFF_WHH0;   // [2048][544]
  const unsigned short* wih1 = ws + OFF_WIH1;   // [2048][512]
  const unsigned short* whh1 = ws + OFF_WHH1;   // [2048][512]
  const unsigned short* wpc  = ws + OFF_WPC;    // [16][512]

  const int tid  = threadIdx.x;
  const int wave = tid >> 5;
  const int lane = tid & 31;
  const int lh   = lane >> 4;
  const int ln   = lane & 15;
  const int ub   = wave * 32;
  const int base = blockIdx.x * 16;

  for (int i = tid; i < 16*AST; i += 512){ h0buf[i]=0; h1buf[i]=0; }
  for (int i = tid; i < FOURH; i += 512){
    lb0[i] = bih0[i] + bhh0[i];
    lb1[i] = bih1[i] + bhh1[i];
  }

  // per-lane weight-row bases (constant column offsets folded into IOFFSET)
  const unsigned nb0 = (unsigned)((ub + ln)*W0C + lh*8);
  const unsigned nb1 = (unsigned)((ub + ln)*HH  + lh*8);
  const int uSt0 = ub + ln, uSt1 = ub + 16 + ln;
  const int un   = ub + ln;                  // hidden-unit index for bias loads
  const int aOff  = ln*AST + lh*8;
  const unsigned pcOff = (unsigned)(ln*HH + lh*8);
  float bpcv = (ln < 3) ? bpc[ln] : 0.0f;
  int slen[8];
  #pragma unroll
  for (int j=0;j<8;++j) slen[j] = (wave==0) ? seqlen[base + j + 8*lh] : 0;

  v8f c0f[2] = {}; v8f c1f[2] = {};
  __syncthreads();

  for (int t = 0; t < TSTEP; ++t){
    v8f acc[8];

    // ===== layer 0: gates = [h0|x] @ Whh0aug^T  (4-deep pipelined B ring) =====
    #pragma unroll
    for (int idx=0; idx<8; ++idx) acc[idx] = (v8f){};
    {
      v16bf bf[4];
      #pragma unroll
      for (int p=0;p<4;++p) bf[p] = ldfrag(whh0, nb0 + GOFF[p]*W0C);
      for (int k=0; k<K0; ++k){
        v16bf afr = ldfrag(&h0buf[aOff + k*32], 0u);
        unsigned vo  = nb0 + (unsigned)k*32u;
        unsigned von = nb0 + (unsigned)((k+1 < K0) ? k+1 : k)*32u;
        #pragma unroll
        for (int idx=0; idx<8; ++idx){
          acc[idx] = __builtin_amdgcn_wmma_f32_16x16x32_bf16(
              false, afr, false, bf[idx&3], (short)0, acc[idx], false, false);
          // prefetch 4 iterations ahead
          unsigned pvo = (idx<4) ? (vo + GOFF[idx+4]*W0C) : (von + GOFF[idx-4]*W0C);
          bf[idx&3] = ldfrag(whh0, pvo);
        }
      }
    }
    __syncthreads();                          // reads of old h0/x done

    #pragma unroll
    for (int t2=0; t2<2; ++t2){
      v8f ai = acc[t2]   + lb0[un + t2*16];
      v8f af = acc[2+t2] + lb0[un + t2*16 + 512];
      v8f ag = acc[4+t2] + lb0[un + t2*16 + 1024];
      v8f ao = acc[6+t2] + lb0[un + t2*16 + 1536];
      #pragma unroll
      for (int j=0;j<8;++j){
        float c = fast_sigm(af[j])*c0f[t2][j] + fast_sigm(ai[j])*fast_tanh(ag[j]);
        c0f[t2][j] = c;
        float h = fast_sigm(ao[j])*fast_tanh(c);
        h0buf[(j + 8*lh)*AST + (t2 ? uSt1 : uSt0)] = f2bf(h);
      }
    }
    __syncthreads();                          // new h0 visible

    // ===== layer 1: gates = h0@Wih1^T + h1@Whh1^T  (two pipelined passes) =====
    #pragma unroll
    for (int idx=0; idx<8; ++idx) acc[idx] = (v8f){};
    {
      v16bf bf[4];
      #pragma unroll
      for (int p=0;p<4;++p) bf[p] = ldfrag(wih1, nb1 + GOFF[p]*HH);
      for (int k=0; k<K1; ++k){
        v16bf afr = ldfrag(&h0buf[aOff + k*32], 0u);
        unsigned vo  = nb1 + (unsigned)k*32u;
        unsigned von = nb1 + (unsigned)((k+1 < K1) ? k+1 : k)*32u;
        #pragma unroll
        for (int idx=0; idx<8; ++idx){
          acc[idx] = __builtin_amdgcn_wmma_f32_16x16x32_bf16(
              false, afr, false, bf[idx&3], (short)0, acc[idx], false, false);
          unsigned pvo = (idx<4) ? (vo + GOFF[idx+4]*HH) : (von + GOFF[idx-4]*HH);
          bf[idx&3] = ldfrag(wih1, pvo);
        }
      }
    }
    {
      v16bf bf[4];
      #pragma unroll
      for (int p=0;p<4;++p) bf[p] = ldfrag(whh1, nb1 + GOFF[p]*HH);
      for (int k=0; k<K1; ++k){
        v16bf afr = ldfrag(&h1buf[aOff + k*32], 0u);
        unsigned vo  = nb1 + (unsigned)k*32u;
        unsigned von = nb1 + (unsigned)((k+1 < K1) ? k+1 : k)*32u;
        #pragma unroll
        for (int idx=0; idx<8; ++idx){
          acc[idx] = __builtin_amdgcn_wmma_f32_16x16x32_bf16(
              false, afr, false, bf[idx&3], (short)0, acc[idx], false, false);
          unsigned pvo = (idx<4) ? (vo + GOFF[idx+4]*HH) : (von + GOFF[idx-4]*HH);
          bf[idx&3] = ldfrag(whh1, pvo);
        }
      }
    }
    __syncthreads();                          // reads of old h1 done

    #pragma unroll
    for (int t2=0; t2<2; ++t2){
      v8f ai = acc[t2]   + lb1[un + t2*16];
      v8f af = acc[2+t2] + lb1[un + t2*16 + 512];
      v8f ag = acc[4+t2] + lb1[un + t2*16 + 1024];
      v8f ao = acc[6+t2] + lb1[un + t2*16 + 1536];
      #pragma unroll
      for (int j=0;j<8;++j){
        float c = fast_sigm(af[j])*c1f[t2][j] + fast_sigm(ai[j])*fast_tanh(ag[j]);
        c1f[t2][j] = c;
        float h = fast_sigm(ao[j])*fast_tanh(c);
        h1buf[(j + 8*lh)*AST + (t2 ? uSt1 : uSt0)] = f2bf(h);
      }
    }
    __syncthreads();                          // new h1 visible

    // ===== projection: point = h1 @ Wpc^T + bpc  (wave 0, ring-2 pipeline) =====
    if (wave == 0){
      v8f p = {};
      v16bf pb = ldfrag(wpc, pcOff);
      for (int k=0; k<K1; ++k){
        v16bf pa = ldfrag(&h1buf[aOff + k*32], 0u);
        unsigned kn = (unsigned)((k+1 < K1) ? k+1 : k);
        v16bf pbn = ldfrag(wpc, pcOff + kn*32u);
        p = __builtin_amdgcn_wmma_f32_16x16x32_bf16(
            false, pa, false, pb, (short)0, p, false, false);
        pb = pbn;
      }
      if (ln < 3){
        #pragma unroll
        for (int j=0;j<8;++j){
          int m = j + 8*lh;
          float val = p[j] + bpcv;
          h0buf[m*AST + HH + ln] = f2bf(val);   // feedback x into augmented cols
          float ov = (t < slen[j]) ? val : 0.0f;
          __builtin_nontemporal_store(ov, &out[((base+m)*TSTEP + t)*3 + ln]);
        }
      }
    }
    __syncthreads();                          // x feedback visible for next step
  }
}

// ---------- launch ----------
extern "C" void kernel_launch(void* const* d_in, const int* in_sizes, int n_in,
                              void* d_out, int out_size, void* d_ws, size_t ws_size,
                              hipStream_t stream){
  // inputs: 0 features(unused) 1 seq_lengths 2 W_ih0 3 W_hh0 4 b_ih0 5 b_hh0
  //         6 W_ih1 7 W_hh1 8 b_ih1 9 b_hh1 10 W_pc 11 b_pc
  const int*   seqlen = (const int*)  d_in[1];
  const float* wih0   = (const float*)d_in[2];
  const float* whh0   = (const float*)d_in[3];
  const float* bih0   = (const float*)d_in[4];
  const float* bhh0   = (const float*)d_in[5];
  const float* wih1   = (const float*)d_in[6];
  const float* whh1   = (const float*)d_in[7];
  const float* bih1   = (const float*)d_in[8];
  const float* bhh1   = (const float*)d_in[9];
  const float* wpc    = (const float*)d_in[10];
  const float* bpc    = (const float*)d_in[11];
  unsigned short* ws  = (unsigned short*)d_ws;   // ~6.2 MB bf16 staging
  float* out          = (float*)d_out;

  prep_weights<<<(WS_ELEMS + 255)/256, 256, 0, stream>>>(whh0, wih0, wih1, whh1, wpc, ws);
  lstm_persist<<<BB/16, 512, 0, stream>>>(seqlen, bih0, bhh0, bih1, bhh1, bpc, ws, out);
}
